// AttentionLS_26723286516317
// MI455X (gfx1250) — compile-verified, matrix operand order checked by reading, not run on hardware
//
#include <hip/hip_runtime.h>

typedef __attribute__((ext_vector_type(16))) _Float16 v16h;
typedef __attribute__((ext_vector_type(8)))  float    v8f;

#define HH    6
#define RR    2
#define WW    8
#define SIDE_ 4
#define TOT_  16
#define DIMC  192
#define DH    32
#define NX_   56
#define NG_   7
#define BB    16
#define NF_   (NX_*NX_)        // 3136
#define NTOK  (NF_+1)          // 3137
#define SCALE_Q 0.17677669529663687f
#define EPS_  1e-5f

// (lane, element) -> K map for 16-bit WMMA A/B operands. Per lane-half this is a
// bijection whose 16 elements form two contiguous 8-wide K runs:
//   e in [0,8)  -> K = half*8 + e
//   e in [8,16) -> K = 16 + half*8 + (e-8)
__device__ __forceinline__ int wk(int lane, int e) {
    int v = e >> 1;
    return ((v >> 2) << 4) | ((lane >> 4) << 3) | ((v & 3) << 1) | (e & 1);
}

__device__ __forceinline__ bool kvalid(int g1, int g2, int kk) {
    int u1 = kk >> 4, u2 = kk & 15;
    return (g1 > 0 || u1 >= SIDE_) && (g1 < NG_ - 1 || u1 < TOT_ - SIDE_)
        && (g2 > 0 || u2 >= SIDE_) && (g2 < NG_ - 1 || u2 < TOT_ - SIDE_);
}

// Build a fragment (16 contiguous halfs) from two contiguous 8-float runs.
__device__ __forceinline__ void store_frag_2run(_Float16* dst,
                                                const float* s0, const float* s1) {
    v16h f;
    #pragma unroll
    for (int j = 0; j < 8; ++j) f[j] = (_Float16)s0[j];
    #pragma unroll
    for (int j = 0; j < 8; ++j) f[8 + j] = (_Float16)s1[j];
    *(v16h*)dst = f;
}

// ------------------------------------------- K0: weights -> fragment-major f16
// wqf: frag id = nt*6 + kc (nt<36), layout [frag][lane][e]; wpf same with nt<12.
__global__ __launch_bounds__(256) void k_convert_w(const float* __restrict__ Wqkv,
                                                   const float* __restrict__ Wproj,
                                                   _Float16* __restrict__ wqf,
                                                   _Float16* __restrict__ wpf) {
    int idx = blockIdx.x * 256 + threadIdx.x;
    if (idx < DIMC * 3 * DIMC) {
        int e = idx & 15, lane = (idx >> 4) & 31, f = idx >> 9;
        int kc = f % 6, nt = f / 6;
        int K = kc * 32 + wk(lane, e);
        wqf[idx] = (_Float16)Wqkv[K * (3 * DIMC) + nt * 16 + (lane & 15)];
    }
    if (idx < DIMC * DIMC) {
        int e = idx & 15, lane = (idx >> 4) & 31, f = idx >> 9;
        int kc = f % 6, nt = f / 6;
        int K = kc * 32 + wk(lane, e);
        wpf[idx] = (_Float16)Wproj[K * DIMC + nt * 16 + (lane & 15)];
    }
}

// ------------------------------------------- K1: qkv GEMM + q-scale + LN(k,v) + dp logits
__global__ __launch_bounds__(128) void k_qkv(
    const float* __restrict__ x, const _Float16* __restrict__ wqf,
    const float* __restrict__ lng, const float* __restrict__ lnb,
    const float* __restrict__ Wdp, const float* __restrict__ bdp,
    float* __restrict__ qbuf, float* __restrict__ kbuf, float* __restrict__ vbuf,
    float* __restrict__ qcls, float* __restrict__ kcls, float* __restrict__ vcls,
    float* __restrict__ dplog)
{
    __shared__ float sx[16][DIMC + 1];
    __shared__ float sqkv[16][3 * DIMC + 1];
    __shared__ _Float16 saf[6 * 32 * 16];        // A fragments: [kc][lane][e]
    __shared__ float sred[16][8];
    __shared__ float smv[16][2];
    const int tid = threadIdx.x;
    const int lane = tid & 31;
    const int wv = tid >> 5;
    const int base = blockIdx.x * 16;

    for (int idx = tid; idx < 16 * DIMC; idx += 128) {
        int r = idx / DIMC, c = idx % DIMC;
        sx[r][c] = x[(long)(base + r) * DIMC + c];
    }
    __syncthreads();

    // build A fragments from LDS f32 (two 32B runs per fragment)
    for (int t = tid; t < 16 * 2 * 6; t += 128) {
        int row = t / 12, rem = t % 12, kc = rem >> 1, half = rem & 1;
        const float* s0 = &sx[row][kc * 32 + half * 8];
        store_frag_2run(&saf[(kc * 32 + (half << 4) + row) * 16], s0, s0 + 16);
    }
    __syncthreads();

    // 16 x 576 GEMM tile: 4 waves x 9 N-tiles, K fully unrolled (6 x 32)
    for (int nt = wv * 9; nt < wv * 9 + 9; ++nt) {
        const _Float16* wptr = wqf + (size_t)nt * 6 * 32 * 16;
        if (nt + 1 < 36) __builtin_prefetch(wqf + (size_t)(nt + 1) * 6 * 32 * 16, 0, 1);
        v8f acc = {};
        #pragma unroll
        for (int kc = 0; kc < 6; ++kc) {
            v16h a  = *(const v16h*)(saf + (kc * 32 + lane) * 16);
            v16h bf = *(const v16h*)(wptr + (kc * 32 + lane) * 16);
            acc = __builtin_amdgcn_wmma_f32_16x16x32_f16(false, a, false, bf,
                                                         (short)0, acc, false, false);
        }
        #pragma unroll
        for (int r = 0; r < 8; ++r) {
            int m = ((lane >> 4) << 3) + r;
            sqkv[m][nt * 16 + (lane & 15)] = acc[r];
        }
    }
    __syncthreads();

    // epilogue: 8 threads per row
    {
        const int row = tid >> 3, sub = tid & 7;
        const int token = base + row;
        const int b = token / NTOK;
        const int n = token % NTOK;
        const int p = n - 1;
        for (int c = sub; c < DIMC; c += 8) {
            float val = sqkv[row][c] * SCALE_Q;
            int h = c >> 5, dd = c & 31;
            if (n == 0) qcls[(b * HH + h) * DH + dd] = val;
            else        qbuf[((long)(b * HH + h) * NF_ + p) * DH + dd] = val;
        }
        for (int part = 1; part <= 2; ++part) {
            float s = 0.f;
            for (int c = sub; c < DIMC; c += 8) s += sqkv[row][part * DIMC + c];
            sred[row][sub] = s; __syncthreads();
            if (sub == 0) {
                float tot = 0.f;
                for (int j = 0; j < 8; ++j) tot += sred[row][j];
                smv[row][0] = tot / (float)DIMC;
            }
            __syncthreads();
            float m = smv[row][0];
            float s2 = 0.f;
            for (int c = sub; c < DIMC; c += 8) {
                float d0 = sqkv[row][part * DIMC + c] - m; s2 += d0 * d0;
            }
            sred[row][sub] = s2; __syncthreads();
            if (sub == 0) {
                float tot = 0.f;
                for (int j = 0; j < 8; ++j) tot += sred[row][j];
                smv[row][1] = rsqrtf(tot / (float)DIMC + EPS_);
            }
            __syncthreads();
            float inv = smv[row][1];
            float* dstb = (part == 1) ? kbuf : vbuf;
            float* dstc = (part == 1) ? kcls : vcls;
            for (int c = sub; c < DIMC; c += 8) {
                float val = (sqkv[row][part * DIMC + c] - m) * inv * lng[c] + lnb[c];
                int h = c >> 5, dd = c & 31;
                if (n == 0) dstc[(b * HH + h) * DH + dd] = val;
                else        dstb[((long)(b * HH + h) * NF_ + p) * DH + dd] = val;
            }
            __syncthreads();
        }
    }
    // dynamic-projection logits
    for (int idx = tid; idx < 16 * 12; idx += 128) {
        int r = idx / 12, c12 = idx % 12;
        int token = base + r;
        int b = token / NTOK, n = token % NTOK;
        if (n == 0) continue;
        int p = n - 1;
        float acc = bdp[c12];
        for (int k = 0; k < DIMC; ++k) acc += sx[r][k] * Wdp[k * 12 + c12];
        dplog[((long)(b * 12 + c12)) * NF_ + p] = acc;
    }
}

// ---------------------------------------------------------------- K2a: softmax over tokens
__global__ __launch_bounds__(256) void k_dpsoftmax(const float* __restrict__ dplog,
                                                   float* __restrict__ cbuf) {
    __shared__ float red[256];
    const int t = threadIdx.x;
    const float* src = dplog + (long)blockIdx.x * NF_;
    float mx = -1e30f;
    for (int p = t; p < NF_; p += 256) mx = fmaxf(mx, src[p]);
    red[t] = mx; __syncthreads();
    for (int s = 128; s > 0; s >>= 1) { if (t < s) red[t] = fmaxf(red[t], red[t + s]); __syncthreads(); }
    mx = red[0]; __syncthreads();
    float sm = 0.f;
    for (int p = t; p < NF_; p += 256) sm += __expf(src[p] - mx);
    red[t] = sm; __syncthreads();
    for (int s = 128; s > 0; s >>= 1) { if (t < s) red[t] += red[t + s]; __syncthreads(); }
    float inv = 1.f / red[0];
    float* dst = cbuf + (long)blockIdx.x * NF_;
    for (int p = t; p < NF_; p += 256) dst[p] = __expf(src[p] - mx) * inv;
}

// ---------------------------------------------------------------- K2b: landmarks c@k, c@v
__global__ __launch_bounds__(256) void k_lms_reduce(const float* __restrict__ cbuf,
                                                    const float* __restrict__ kbuf,
                                                    const float* __restrict__ vbuf,
                                                    float* __restrict__ klms,
                                                    float* __restrict__ vlms) {
    __shared__ float red[8][DH];
    const int sel = blockIdx.x & 1;
    const int idx = blockIdx.x >> 1;
    const int b = idx / (HH * RR);
    const int rem = idx % (HH * RR);
    const int h = rem / RR, r = rem % RR;
    const int c12 = h * RR + r;
    const float* src = (sel == 0 ? kbuf : vbuf) + ((long)(b * HH + h)) * NF_ * DH;
    const float* cc = cbuf + ((long)(b * 12 + c12)) * NF_;
    const int g = threadIdx.x >> 5, dd = threadIdx.x & 31;
    float acc = 0.f;
    for (int p = g; p < NF_; p += 8) acc += cc[p] * src[(long)p * DH + dd];
    red[g][dd] = acc; __syncthreads();
    if (g == 0) {
        for (int j = 1; j < 8; ++j) acc += red[j][dd];
        float* dst = (sel == 0 ? klms : vlms);
        dst[(b * RR + r) * DIMC + h * DH + dd] = acc;
    }
}

// ---------------------------------------------------------------- K2c: LN on landmarks
__global__ __launch_bounds__(256) void k_lms_ln(float* __restrict__ klms,
                                                float* __restrict__ vlms,
                                                const float* __restrict__ g,
                                                const float* __restrict__ bta) {
    __shared__ float red[256];
    const int t = threadIdx.x;
    const int sel = blockIdx.x & 1;
    const int row = blockIdx.x >> 1;
    float* buf = (sel == 0 ? klms : vlms) + row * DIMC;
    float val = (t < DIMC) ? buf[t] : 0.f;
    red[t] = val; __syncthreads();
    for (int s = 128; s > 0; s >>= 1) { if (t < s) red[t] += red[t + s]; __syncthreads(); }
    float m = red[0] / (float)DIMC; __syncthreads();
    float d0 = (t < DIMC) ? (val - m) : 0.f;
    red[t] = d0 * d0; __syncthreads();
    for (int s = 128; s > 0; s >>= 1) { if (t < s) red[t] += red[t + s]; __syncthreads(); }
    float inv = rsqrtf(red[0] / (float)DIMC + EPS_);
    if (t < DIMC) buf[t] = (val - m) * inv * g[t] + bta[t];
}

// ---------------------------------------------------------------- K3: windowed attention tile
__global__ __launch_bounds__(128) void k_attn(
    const float* __restrict__ qbuf, const float* __restrict__ kbuf,
    const float* __restrict__ vbuf, const float* __restrict__ klms,
    const float* __restrict__ vlms, const float* __restrict__ kcls,
    const float* __restrict__ vcls, float* __restrict__ aout)
{
    extern __shared__ char smem[];
    _Float16* skf  = (_Float16*)smem;          // 16 frags  (dots B): 16*32*16
    _Float16* svf  = skf + 16 * 32 * 16;       // 16 frags  (PV B):   8*2*32*16
    _Float16* sqa  = svf + 16 * 32 * 16;       // 4 frags   (dots A): 4*32*16
    _Float16* satf = sqa + 4 * 32 * 16;        // 32 frags  (PV A):   4*8*32*16
    float*    sq   = (float*)(satf + 32 * 32 * 16);   // 64*32 f32
    float*    sdt  = sq + 64 * DH;                    // 64*260 f32
    float*    sex  = sdt + 64 * 260;
    float*    s_klms = sex;            // 2*32
    float*    s_vlms = sex + 64;       // 2*32
    float*    s_kcls = sex + 128;      // 32
    float*    s_vcls = sex + 160;      // 32
    float*    s_adp  = sex + 192;      // 64*2
    float*    s_acls = sex + 320;      // 64

    const int bx = blockIdx.x;
    const int g2 = bx % NG_;
    const int g1 = (bx / NG_) % NG_;
    const int h  = (bx / (NG_ * NG_)) % HH;
    const int b  = bx / (NG_ * NG_ * HH);
    const int tid = threadIdx.x, lane = tid & 31, wv = tid >> 5;

    const float* qb = qbuf + ((long)(b * HH + h)) * NF_ * DH;
    const float* kb = kbuf + ((long)(b * HH + h)) * NF_ * DH;
    const float* vb = vbuf + ((long)(b * HH + h)) * NF_ * DH;

    // q tile (f32, for the dp/cls dot products)
    for (int idx = tid; idx < 64 * DH; idx += 128) {
        int qq = idx >> 5, dd = idx & 31;
        int p = (g1 * WW + (qq >> 3)) * NX_ + g2 * WW + (qq & 7);
        sq[qq * DH + dd] = qb[(long)p * DH + dd];
    }
    // q A-fragments (direct from global; two 32B runs per fragment)
    for (int t = tid; t < 4 * 32; t += 128) {
        int lane2 = t & 31, mt = t >> 5;
        int row15 = lane2 & 15, half = lane2 >> 4;
        int qq = mt * 16 + row15;
        int p = (g1 * WW + (qq >> 3)) * NX_ + g2 * WW + (qq & 7);
        const float* s0 = qb + (long)p * DH + half * 8;
        store_frag_2run(&sqa[t * 16], s0, s0 + 16);
    }
    // k B-fragments: frag (nt,lane) holds key row nt*16+(lane&15), one lane-half of K
    for (int t = tid; t < 16 * 32; t += 128) {
        int lane2 = t & 31, nt = t >> 5;
        int key = nt * 16 + (lane2 & 15), half = lane2 >> 4;
        int i = g1 * WW + (key >> 4) - SIDE_;
        int j = g2 * WW + (key & 15) - SIDE_;
        bool ok = (i >= 0 && i < NX_ && j >= 0 && j < NX_);
        if (ok) {
            const float* s0 = kb + (long)(i * NX_ + j) * DH + half * 8;
            store_frag_2run(&skf[t * 16], s0, s0 + 16);
        } else {
            v16h z = {};
            *(v16h*)(skf + t * 16) = z;
        }
    }
    // v B-fragments for PV: slot (lane,e) -> key kc*32+wk(lane,e), dim nh*16+(lane&15)
    for (int t = tid; t < 8 * 2 * 32; t += 128) {
        int lane2 = t & 31, nh = (t >> 5) & 1, kc = t >> 6;
        int dim = nh * 16 + (lane2 & 15);
        v16h f;
        #pragma unroll
        for (int e = 0; e < 16; ++e) {
            int key = kc * 32 + wk(lane2, e);
            int i = g1 * WW + (key >> 4) - SIDE_;
            int j = g2 * WW + (key & 15) - SIDE_;
            bool ok = (i >= 0 && i < NX_ && j >= 0 && j < NX_);
            f[e] = ok ? (_Float16)vb[(long)(i * NX_ + j) * DH + dim] : (_Float16)0.f;
        }
        *(v16h*)(svf + t * 16) = f;
    }
    if (tid < 64) {
        int r = tid >> 5, dd = tid & 31;
        s_klms[r * DH + dd] = klms[(b * RR + r) * DIMC + h * DH + dd];
        s_vlms[r * DH + dd] = vlms[(b * RR + r) * DIMC + h * DH + dd];
    } else if (tid < 96) {
        int dd = tid & 31;
        s_kcls[dd] = kcls[(b * HH + h) * DH + dd];
        s_vcls[dd] = vcls[(b * HH + h) * DH + dd];
    }
    __syncthreads();

    // dots = q (64x32) @ k^T (32x256): contiguous fragment loads + one WMMA per N-tile
    {
        v16h a = *(const v16h*)(sqa + (wv * 32 + lane) * 16);
        for (int nt = 0; nt < 16; ++nt) {
            v16h bf = *(const v16h*)(skf + (nt * 32 + lane) * 16);
            v8f acc = {};
            acc = __builtin_amdgcn_wmma_f32_16x16x32_f16(false, a, false, bf,
                                                         (short)0, acc, false, false);
            #pragma unroll
            for (int r = 0; r < 8; ++r)
                sdt[(wv * 16 + ((lane >> 4) << 3) + r) * 260 + nt * 16 + (lane & 15)] = acc[r];
        }
    }
    if (tid < 64) {
        float d0 = 0.f, d1 = 0.f, dc = 0.f;
        for (int dd = 0; dd < DH; ++dd) {
            float qv = sq[tid * DH + dd];
            d0 += qv * s_klms[dd];
            d1 += qv * s_klms[DH + dd];
            dc += qv * s_kcls[dd];
        }
        sdt[tid * 260 + 256] = d0;
        sdt[tid * 260 + 257] = d1;
        sdt[tid * 260 + 258] = dc;
    }
    __syncthreads();

    // softmax over [2 dp, 256 windowed (masked), 1 cls]; write PV A-fragments directly
    if (tid < 64) {
        const float* row = sdt + tid * 260;
        float mx = fmaxf(fmaxf(row[256], row[257]), row[258]);
        for (int kk = 0; kk < 256; ++kk)
            if (kvalid(g1, g2, kk)) mx = fmaxf(mx, row[kk]);
        float sm = __expf(row[256] - mx) + __expf(row[257] - mx) + __expf(row[258] - mx);
        for (int kk = 0; kk < 256; ++kk)
            if (kvalid(g1, g2, kk)) sm += __expf(row[kk] - mx);
        float inv = 1.f / sm;
        int mt = tid >> 4, m15 = tid & 15;
        for (int kk = 0; kk < 256; ++kk) {
            float val = kvalid(g1, g2, kk) ? __expf(row[kk] - mx) * inv : 0.f;
            int kc = kk >> 5, K = kk & 31;
            int half = (K >> 3) & 1, g = K >> 4, r = K & 7;
            int e = ((((g << 2) | (r >> 1)) << 1) | (K & 1));
            satf[(((mt * 8 + kc) * 32) + (half << 4) + m15) * 16 + e] = (_Float16)val;
        }
        s_adp[tid * 2 + 0] = __expf(row[256] - mx) * inv;
        s_adp[tid * 2 + 1] = __expf(row[257] - mx) * inv;
        s_acls[tid]        = __expf(row[258] - mx) * inv;
    }
    __syncthreads();

    // out = attn (64x256) @ v (256x32): all-contiguous fragment loads, 2 WMMAs per chunk
    {
        v8f acc0 = {}, acc1 = {};
        #pragma unroll
        for (int kc = 0; kc < 8; ++kc) {
            v16h a  = *(const v16h*)(satf + ((wv * 8 + kc) * 32 + lane) * 16);
            v16h b0 = *(const v16h*)(svf + ((kc * 2 + 0) * 32 + lane) * 16);
            v16h b1 = *(const v16h*)(svf + ((kc * 2 + 1) * 32 + lane) * 16);
            acc0 = __builtin_amdgcn_wmma_f32_16x16x32_f16(false, a, false, b0,
                                                          (short)0, acc0, false, false);
            acc1 = __builtin_amdgcn_wmma_f32_16x16x32_f16(false, a, false, b1,
                                                          (short)0, acc1, false, false);
        }
        #pragma unroll
        for (int r = 0; r < 8; ++r) {
            int qq = wv * 16 + ((lane >> 4) << 3) + r;
            int p = (g1 * WW + (qq >> 3)) * NX_ + g2 * WW + (qq & 7);
            float* dst = aout + ((long)b * NF_ + p) * DIMC + h * DH;
            int n0 = lane & 15, n1 = 16 + (lane & 15);
            float c0 = s_adp[qq * 2] * s_vlms[n0] + s_adp[qq * 2 + 1] * s_vlms[DH + n0]
                     + s_acls[qq] * s_vcls[n0];
            float c1 = s_adp[qq * 2] * s_vlms[n1] + s_adp[qq * 2 + 1] * s_vlms[DH + n1]
                     + s_acls[qq] * s_vcls[n1];
            dst[n0] = acc0[r] + c0;
            dst[n1] = acc1[r] + c1;
        }
    }
}

// ---------------------------------------------------------------- K4: cls-token attention
__global__ __launch_bounds__(256) void k_cls(
    const float* __restrict__ qcls, const float* __restrict__ kcls,
    const float* __restrict__ vcls, const float* __restrict__ aout,
    float* __restrict__ clsn)
{
    __shared__ float red[256];
    __shared__ float sqc[DH];
    __shared__ float racc[256][DH];
    const int t = threadIdx.x;
    const int h = blockIdx.x % HH, b = blockIdx.x / HH;
    if (t < DH) sqc[t] = qcls[(b * HH + h) * DH + t];
    __syncthreads();
    const float* ab = aout + (long)b * NF_ * DIMC + h * DH;
    float l0 = 0.f;
    for (int dd = 0; dd < DH; ++dd) l0 += sqc[dd] * kcls[(b * HH + h) * DH + dd];
    float mx = l0;
    for (int p = t; p < NF_; p += 256) {
        float l = 0.f;
        for (int dd = 0; dd < DH; ++dd) l += sqc[dd] * ab[(long)p * DIMC + dd];
        mx = fmaxf(mx, l);
    }
    red[t] = mx; __syncthreads();
    for (int s = 128; s > 0; s >>= 1) { if (t < s) red[t] = fmaxf(red[t], red[t + s]); __syncthreads(); }
    mx = red[0]; __syncthreads();
    float sm = 0.f;
    float acc[DH];
    for (int dd = 0; dd < DH; ++dd) acc[dd] = 0.f;
    for (int p = t; p < NF_; p += 256) {
        float l = 0.f;
        for (int dd = 0; dd < DH; ++dd) l += sqc[dd] * ab[(long)p * DIMC + dd];
        float w = __expf(l - mx);
        sm += w;
        for (int dd = 0; dd < DH; ++dd) acc[dd] += w * ab[(long)p * DIMC + dd];
    }
    red[t] = sm; __syncthreads();
    for (int s = 128; s > 0; s >>= 1) { if (t < s) red[t] += red[t + s]; __syncthreads(); }
    float tot = red[0] + __expf(l0 - mx);
    for (int dd = 0; dd < DH; ++dd) racc[t][dd] = acc[dd];
    __syncthreads();
    for (int s = 128; s > 0; s >>= 1) {
        if (t < s) for (int dd = 0; dd < DH; ++dd) racc[t][dd] += racc[t + s][dd];
        __syncthreads();
    }
    if (t < DH)
        clsn[b * DIMC + h * DH + t] =
            (racc[0][t] + __expf(l0 - mx) * vcls[(b * HH + h) * DH + t]) / tot;
}

// ---------------------------------------------------------------- K5: output projection
__global__ __launch_bounds__(128) void k_proj(
    const float* __restrict__ aout, const float* __restrict__ clsn,
    const _Float16* __restrict__ wpf, const float* __restrict__ bproj,
    float* __restrict__ out)
{
    __shared__ _Float16 saf[6 * 32 * 16];      // A fragments [kc][lane][e]
    const int tid = threadIdx.x;
    const int lane = tid & 31;
    const int wv = tid >> 5;
    const int base = blockIdx.x * 16;
    // build A fragments directly from global rows (two 32B runs each)
    for (int t = tid; t < 16 * 2 * 6; t += 128) {
        int row = t / 12, rem = t % 12, kc = rem >> 1, half = rem & 1;
        int token = base + row;
        int b = token / NTOK, n = token % NTOK;
        const float* rp = (n == 0) ? (clsn + b * DIMC)
                                   : (aout + ((long)b * NF_ + (n - 1)) * DIMC);
        const float* s0 = rp + kc * 32 + half * 8;
        store_frag_2run(&saf[(kc * 32 + (half << 4) + row) * 16], s0, s0 + 16);
    }
    __syncthreads();
    for (int nt = wv * 3; nt < wv * 3 + 3; ++nt) {
        const _Float16* wptr = wpf + (size_t)nt * 6 * 32 * 16;
        if (nt + 1 < 12) __builtin_prefetch(wpf + (size_t)(nt + 1) * 6 * 32 * 16, 0, 1);
        v8f acc = {};
        #pragma unroll
        for (int kc = 0; kc < 6; ++kc) {
            v16h a  = *(const v16h*)(saf + (kc * 32 + lane) * 16);
            v16h bf = *(const v16h*)(wptr + (kc * 32 + lane) * 16);
            acc = __builtin_amdgcn_wmma_f32_16x16x32_f16(false, a, false, bf,
                                                         (short)0, acc, false, false);
        }
        #pragma unroll
        for (int r = 0; r < 8; ++r) {
            int m = ((lane >> 4) << 3) + r;
            int token = base + m;
            int col = nt * 16 + (lane & 15);
            out[(long)token * DIMC + col] = acc[r] + bproj[col];
        }
    }
}

// ---------------------------------------------------------------- launch
extern "C" void kernel_launch(void* const* d_in, const int* in_sizes, int n_in,
                              void* d_out, int out_size, void* d_ws, size_t ws_size,
                              hipStream_t stream) {
    (void)in_sizes; (void)n_in; (void)out_size; (void)ws_size;
    const float* x     = (const float*)d_in[0];
    const float* Wqkv  = (const float*)d_in[1];
    const float* lng   = (const float*)d_in[2];
    const float* lnb   = (const float*)d_in[3];
    const float* Wdp   = (const float*)d_in[4];
    const float* bdp   = (const float*)d_in[5];
    const float* lndg  = (const float*)d_in[6];
    const float* lndb  = (const float*)d_in[7];
    const float* Wproj = (const float*)d_in[8];
    const float* bproj = (const float*)d_in[9];

    float* ws = (float*)d_ws;
    const size_t QSP = (size_t)BB * HH * NF_ * DH;
    float* qbuf  = ws;
    float* kbuf  = qbuf + QSP;
    float* vbuf  = kbuf + QSP;
    float* aout  = vbuf + QSP;
    float* dplog = aout + (size_t)BB * NF_ * DIMC;
    float* cbuf  = dplog + (size_t)BB * 12 * NF_;
    float* klms  = cbuf + (size_t)BB * 12 * NF_;
    float* vlms  = klms + BB * RR * DIMC;
    float* qcls  = vlms + BB * RR * DIMC;
    float* kcls  = qcls + BB * HH * DH;
    float* vcls  = kcls + BB * HH * DH;
    float* clsn  = vcls + BB * HH * DH;
    _Float16* wqf = (_Float16*)(clsn + BB * DIMC);
    _Float16* wpf = wqf + DIMC * 3 * DIMC;

    k_convert_w<<<(DIMC * 3 * DIMC + 255) / 256, 256, 0, stream>>>(Wqkv, Wproj, wqf, wpf);
    k_qkv<<<(BB * NTOK) / 16, 128, 0, stream>>>(x, wqf, lng, lnb, Wdp, bdp,
                                                qbuf, kbuf, vbuf, qcls, kcls, vcls, dplog);
    k_dpsoftmax<<<BB * 12, 256, 0, stream>>>(dplog, cbuf);
    k_lms_reduce<<<BB * HH * RR * 2, 256, 0, stream>>>(cbuf, kbuf, vbuf, klms, vlms);
    k_lms_ln<<<BB * RR * 2, 256, 0, stream>>>(klms, vlms, lndg, lndb);
    const size_t smem3 = (size_t)((16 * 32 * 16 + 16 * 32 * 16 + 4 * 32 * 16
                                   + 32 * 32 * 16) * 2                 // f16 fragments
                                  + (64 * DH + 64 * 260 + 384) * 4);   // f32 regions
    k_attn<<<BB * HH * NG_ * NG_, 128, smem3, stream>>>(qbuf, kbuf, vbuf, klms, vlms,
                                                        kcls, vcls, aout);
    k_cls<<<BB * HH, 256, 0, stream>>>(qcls, kcls, vcls, aout, clsn);
    k_proj<<<(BB * NTOK) / 16, 128, 0, stream>>>(aout, clsn, wpf, bproj, (float*)d_out);
}
